// MoECNBlock_31705448579441
// MI455X (gfx1250) — matrix operand order repaired and checked by reading, not compiled, not run on hardware
//
#include <hip/hip_runtime.h>
#include <math.h>
#include <stdint.h>

#define DIM   384
#define NE    8
#define HID   1536
#define NB    32
#define HH    32
#define WW    32
#define T_TOK (NB*HH*WW)       // 32768 tokens
#define EPS   1e-6f
#define LIST_CAP 32768          // max tokens per expert (top-2 indices distinct)
#define MAX_TILES 4360          // sum_e ceil(cnt_e/16) <= 4096+8

typedef __attribute__((ext_vector_type(16))) __bf16 v16bf;
typedef __attribute__((ext_vector_type(8)))  __bf16 v8bf;
typedef __attribute__((ext_vector_type(8)))  float  v8f;

// ---------------- fragment loaders (CDNA5 wave32 WMMA bf16 layouts) -------------
// A (16x32 bf16): lane L holds row L%16; lanes 0-15: K = kb+{0..7, 16..23},
// lanes 16-31: K = kb+{8..15, 24..31}.  Two contiguous 16B chunks per lane.
__device__ __forceinline__ v16bf frag_a(const __bf16* base, int lane, int kb, int rs) {
  int row = lane & 15;
  int hi  = (lane >> 4) & 1;
  const __bf16* p0 = base + (size_t)row * rs + kb + hi * 8;
  const __bf16* p1 = base + (size_t)row * rs + kb + 16 + hi * 8;
  v8bf lo = *(const v8bf*)p0;
  v8bf hh = *(const v8bf*)p1;
  return __builtin_shufflevector(lo, hh, 0,1,2,3,4,5,6,7,8,9,10,11,12,13,14,15);
}
// B (32x16 bf16), column-major weights (wt[col][k]): lane L holds col L%16;
// lanes 0-15: K = kb+{0..7, 16..23}, lanes 16-31: K = kb+{8..15, 24..31}.  Two 16B chunks per lane.
__device__ __forceinline__ v16bf frag_b(const __bf16* colbase, int lane, int kb) {
  int hi = (lane >> 4) & 1;
  const __bf16* p = colbase + kb + hi * 8;
  v8bf lo = *(const v8bf*)p;
  v8bf hh = *(const v8bf*)(p + 16);
  return __builtin_shufflevector(lo, hh, 0,1,2,3,4,5,6,7,8,9,10,11,12,13,14,15);
}

__device__ __forceinline__ float gelu_exact(float x) {
  return 0.5f * x * (1.0f + erff(x * 0.70710678118654752f));
}

__device__ __forceinline__ void wait_asynccnt0() {
#if __has_builtin(__builtin_amdgcn_s_wait_asynccnt)
  __builtin_amdgcn_s_wait_asynccnt(0);
#else
  asm volatile("s_wait_asynccnt 0" ::: "memory");
#endif
}

// ---------------- 1) depthwise 7x7 conv, NCHW in -> NHWC f32 out ----------------
__global__ void dwconv_kernel(const float* __restrict__ in, const float* __restrict__ w,
                              const float* __restrict__ b, float* __restrict__ xc) {
  int nc = blockIdx.x;
  int n = nc / DIM, c = nc % DIM;
  __shared__ float tile[38 * 38];
  __shared__ float wsh[49];
  int tid = threadIdx.x;
  const float* plane = in + ((size_t)(n * DIM + c)) * (HH * WW);
  for (int i = tid; i < 38 * 38; i += blockDim.x) {
    int ih = i / 38 - 3, iw = i % 38 - 3;
    float v = 0.0f;
    if (ih >= 0 && ih < HH && iw >= 0 && iw < WW) v = plane[ih * WW + iw];
    tile[i] = v;
  }
  if (tid < 49) wsh[tid] = w[c * 49 + tid];
  __syncthreads();
  float bias = b[c];
  for (int p = tid; p < HH * WW; p += blockDim.x) {
    int h = p / WW, wq = p % WW;
    float acc = bias;
#pragma unroll
    for (int kh = 0; kh < 7; ++kh)
#pragma unroll
      for (int kw = 0; kw < 7; ++kw)
        acc += tile[(h + kh) * 38 + (wq + kw)] * wsh[kh * 7 + kw];
    xc[((size_t)(n * HH * WW + p)) * DIM + c] = acc;
  }
}

// ---------------- 2) zero routing metadata ----------------
__global__ void init_meta_kernel(int* meta) {
  if (threadIdx.x < 16) meta[threadIdx.x] = 0;
}

// ---------------- 3) LayerNorm + gating top-2 + bf16 activation -----------------
__global__ void ln_route_kernel(const float* __restrict__ xc, const float* __restrict__ g,
                                const float* __restrict__ bta, const float* __restrict__ gate_w,
                                __bf16* __restrict__ xn, int* counts, int* lists, float* gates) {
  int t = blockIdx.x;
  int tid = threadIdx.x;  // 128 threads, 3 channels each
  __shared__ float red[128];
  __shared__ float lg[NE];
  const float* row = xc + (size_t)t * DIM;
  float v[3];
#pragma unroll
  for (int i = 0; i < 3; i++) v[i] = row[tid + i * 128];
  float s = v[0] + v[1] + v[2];
  red[tid] = s; __syncthreads();
  for (int st = 64; st > 0; st >>= 1) { if (tid < st) red[tid] += red[tid + st]; __syncthreads(); }
  float mu = red[0] * (1.0f / DIM); __syncthreads();
  float sq = 0.0f;
#pragma unroll
  for (int i = 0; i < 3; i++) { float d = v[i] - mu; sq += d * d; }
  red[tid] = sq; __syncthreads();
  for (int st = 64; st > 0; st >>= 1) { if (tid < st) red[tid] += red[tid + st]; __syncthreads(); }
  float rstd = rsqrtf(red[0] * (1.0f / DIM) + EPS); __syncthreads();

  float acc[NE];
#pragma unroll
  for (int e = 0; e < NE; e++) acc[e] = 0.0f;
#pragma unroll
  for (int i = 0; i < 3; i++) {
    int c = tid + i * 128;
    float xv = (v[i] - mu) * rstd * g[c] + bta[c];
    xn[(size_t)t * DIM + c] = (__bf16)xv;
#pragma unroll
    for (int e = 0; e < NE; e++) acc[e] += xv * gate_w[e * DIM + c];
  }
  for (int e = 0; e < NE; e++) {
    red[tid] = acc[e]; __syncthreads();
    for (int st = 64; st > 0; st >>= 1) { if (tid < st) red[tid] += red[tid + st]; __syncthreads(); }
    if (tid == 0) lg[e] = red[0];
    __syncthreads();
  }
  if (tid == 0) {
    int i0 = 0; float v0 = lg[0];
    for (int e = 1; e < NE; e++) if (lg[e] > v0) { v0 = lg[e]; i0 = e; }
    int i1 = (i0 == 0) ? 1 : 0; float v1 = lg[i1];
    for (int e = 0; e < NE; e++) if (e != i0 && lg[e] > v1) { v1 = lg[e]; i1 = e; }
    float e1 = expf(v1 - v0);
    float inv = 1.0f / (1.0f + e1);
    float w0 = inv, w1 = e1 * inv;  // softmax over [v0, v1]
    int p0 = atomicAdd(&counts[i0], 1);
    lists[i0 * LIST_CAP + p0] = t; gates[i0 * LIST_CAP + p0] = w0;
    int p1 = atomicAdd(&counts[i1], 1);
    lists[i1 * LIST_CAP + p1] = t; gates[i1 * LIST_CAP + p1] = w1;
  }
}

// ---------------- 4) f32 -> bf16 transposed weights (column-major B) ------------
__global__ void convert_w_kernel(const float* __restrict__ w1, const float* __restrict__ w2,
                                 __bf16* __restrict__ w1t, __bf16* __restrict__ w2t) {
  size_t idx = (size_t)blockIdx.x * blockDim.x + threadIdx.x;
  size_t total = (size_t)NE * DIM * HID;
  if (idx >= total) return;
  int e = (int)(idx / ((size_t)DIM * HID));
  int r = (int)(idx % ((size_t)DIM * HID));
  // w1t[e][nh][k] = w1[e][k][nh]   (w1: [NE][DIM][HID])
  int nh = r / DIM, k = r % DIM;
  w1t[idx] = (__bf16)w1[((size_t)e * DIM + k) * HID + nh];
  // w2t[e][c][k2] = w2[e][k2][c]   (w2: [NE][HID][DIM])
  int c = r / HID, k2 = r % HID;
  w2t[idx] = (__bf16)w2[((size_t)e * HID + k2) * DIM + c];
}

// ---------------- 5) zero accumulator ----------------
__global__ void zero_y_kernel(float4* y) {
  size_t i = (size_t)blockIdx.x * blockDim.x + threadIdx.x;
  if (i < (size_t)T_TOK * DIM / 4) y[i] = make_float4(0.f, 0.f, 0.f, 0.f);
}

// ---------------- 6) build (expert, m-tile) work list ----------------
__global__ void tile_build_kernel(const int* counts, int* total, int* desc) {
  if (threadIdx.x == 0 && blockIdx.x == 0) {
    int tt = 0;
    for (int e = 0; e < NE; e++) {
      int nt = (counts[e] + 15) >> 4;
      for (int i = 0; i < nt && tt < MAX_TILES; i++) { desc[2 * tt] = e; desc[2 * tt + 1] = i; tt++; }
    }
    *total = tt;
  }
}

// ---------------- 7) fused expert FFN: GEMM1 -> GELU -> GEMM2 (WMMA bf16) -------
__global__ __launch_bounds__(128) void moe_kernel(
    const __bf16* __restrict__ xn, const __bf16* __restrict__ w1t, const __bf16* __restrict__ w2t,
    const float* __restrict__ b1, const float* __restrict__ b2,
    const int* __restrict__ counts, const int* __restrict__ total, const int* __restrict__ desc,
    const int* __restrict__ lists, const float* __restrict__ gates, float* __restrict__ y) {
  if ((int)blockIdx.x >= *total) return;
  __shared__ __align__(16) __bf16 a_tile[16][DIM];   // 12 KB
  __shared__ __align__(16) __bf16 h_tile[16][HID];   // 48 KB
  __shared__ int   tok_s[16];
  __shared__ float gate_s[16];
  int tid = threadIdx.x;
  int e    = desc[2 * blockIdx.x];
  int mt   = desc[2 * blockIdx.x + 1];
  int cnt  = counts[e];
  int base = mt * 16;
  if (tid < 16) {
    int gi = base + tid;
    if (gi < cnt) { tok_s[tid] = lists[e * LIST_CAP + gi]; gate_s[tid] = gates[e * LIST_CAP + gi]; }
    else          { tok_s[tid] = -1; gate_s[tid] = 0.0f; }
  }
  __syncthreads();
  // gather 16x384 bf16 A-tile straight into LDS via CDNA5 async DMA
  // (GLOBAL_LOAD_ASYNC_TO_LDS_B128: vdst = LDS byte address VGPR, vaddr = 64-bit
  //  global address, tracked by ASYNCcnt; padding rows zero-filled via ds store)
  for (int ci = tid; ci < 16 * 48; ci += 128) {
    int row = ci / 48, cc = ci % 48;
    int tok = tok_s[row];
    if (tok >= 0) {
      unsigned lds_addr = (unsigned)(uintptr_t)&a_tile[row][cc * 8];
      const __bf16* src = xn + (size_t)tok * DIM + cc * 8;
      asm volatile("global_load_async_to_lds_b128 %0, %1, off"
                   :: "v"(lds_addr), "v"(src)
                   : "memory");
    } else {
      *(uint4*)(&a_tile[row][cc * 8]) = make_uint4(0, 0, 0, 0);
    }
  }
  wait_asynccnt0();   // drain ASYNCcnt before any lane reads the tile
  __syncthreads();

  int lane = tid & 31, wid = tid >> 5;  // 4 waves
  // ---- GEMM1: [16,384] x [384,1536] + b1, GELU -> h_tile bf16 ----
  const __bf16* w1e = w1t + (size_t)e * HID * DIM;
  for (int nt = 0; nt < 24; ++nt) {           // 24 n-tiles of 16 per wave
    int nb  = wid * 384 + nt * 16;
    int col = nb + (lane & 15);
    const __bf16* bcol = w1e + (size_t)col * DIM;
    __builtin_prefetch(bcol, 0, 1);
    v8f acc = {};
    for (int kb = 0; kb < DIM; kb += 32) {
      v16bf a = frag_a(&a_tile[0][0], lane, kb, DIM);
      v16bf b = frag_b(bcol, lane, kb);
      acc = __builtin_amdgcn_wmma_f32_16x16x32_bf16(false, a, false, b, (short)0, acc, false, false);
    }
    float bias = b1[e * HID + col];
    int moff = 8 * (lane >> 4);
#pragma unroll
    for (int r = 0; r < 8; r++) {
      float xv = acc[r] + bias;               // D layout: M = r + 8*(lane>>4), N = lane&15
      h_tile[moff + r][col] = (__bf16)gelu_exact(xv);
    }
  }
  __syncthreads();

  // ---- GEMM2: [16,1536] x [1536,384] + b2, gate-scale, atomic scatter-add ----
  const __bf16* w2e = w2t + (size_t)e * DIM * HID;
  for (int nt = 0; nt < 6; ++nt) {            // 6 n-tiles of 16 per wave
    int nb  = wid * 96 + nt * 16;
    int col = nb + (lane & 15);
    const __bf16* bcol = w2e + (size_t)col * HID;
    __builtin_prefetch(bcol, 0, 1);
    v8f acc = {};
    for (int kb = 0; kb < HID; kb += 32) {
      v16bf a = frag_a(&h_tile[0][0], lane, kb, HID);
      v16bf b = frag_b(bcol, lane, kb);
      acc = __builtin_amdgcn_wmma_f32_16x16x32_bf16(false, a, false, b, (short)0, acc, false, false);
    }
    float b2v = b2[e * DIM + col];
    int moff = 8 * (lane >> 4);
#pragma unroll
    for (int r = 0; r < 8; r++) {
      int M = moff + r;
      int tok = tok_s[M];
      if (tok >= 0) atomicAdd(&y[(size_t)tok * DIM + col], (acc[r] + b2v) * gate_s[M]);
    }
  }
}

// ---------------- 8) residual + layer_scale, NHWC acc -> NCHW out ---------------
__global__ void combine_kernel(const float* __restrict__ in, const float* __restrict__ ls,
                               const float* __restrict__ y, float* __restrict__ out) {
  size_t idx = (size_t)blockIdx.x * blockDim.x + threadIdx.x;
  size_t tot = (size_t)NB * DIM * HH * WW;
  if (idx >= tot) return;
  size_t per_n = (size_t)DIM * HH * WW;
  int n = (int)(idx / per_n);
  size_t rem = idx % per_n;
  int c = (int)(rem / (HH * WW));
  int hw = (int)(rem % (HH * WW));
  size_t t = (size_t)n * (HH * WW) + hw;
  out[idx] = in[idx] + ls[c] * y[t * DIM + c];
}

// ------------------------------- launcher ---------------------------------------
extern "C" void kernel_launch(void* const* d_in, const int* in_sizes, int n_in,
                              void* d_out, int out_size, void* d_ws, size_t ws_size,
                              hipStream_t stream) {
  const float* input   = (const float*)d_in[0];
  const float* dw_w    = (const float*)d_in[1];
  const float* dw_b    = (const float*)d_in[2];
  const float* ln_g    = (const float*)d_in[3];
  const float* ln_b    = (const float*)d_in[4];
  const float* gate_w  = (const float*)d_in[5];
  const float* w1      = (const float*)d_in[6];
  const float* b1      = (const float*)d_in[7];
  const float* w2      = (const float*)d_in[8];
  const float* b2      = (const float*)d_in[9];
  const float* lscale  = (const float*)d_in[10];
  float* out = (float*)d_out;

  // workspace layout (y_acc aliases the conv buffer, dead after ln_route)
  char* w = (char*)d_ws;
  float*  xc    = (float*)w;                                // [T,384] f32, 48 MB
  float*  y_acc = xc;                                       // alias (48 MB)
  __bf16* xn    = (__bf16*)(w + 50331648);                  // [T,384] bf16, 24 MB
  __bf16* w1t   = (__bf16*)(w + 75497472);                  // [8][1536][384] bf16
  __bf16* w2t   = (__bf16*)(w + 84934656);                  // [8][384][1536] bf16
  int*    meta  = (int*)(w + 94371840);                     // counts[8], total at [8]
  int*    counts = meta;
  int*    total  = meta + 8;
  int*    desc   = meta + 64;                               // 2*MAX_TILES ints
  int*    lists  = (int*)(w + 94371840 + 65536);            // [8][32768]
  float*  gates  = (float*)(w + 94371840 + 65536 + 1048576);// [8][32768]

  // 1) depthwise conv -> NHWC
  dwconv_kernel<<<NB * DIM, 256, 0, stream>>>(input, dw_w, dw_b, xc);
  // 2) zero routing meta
  init_meta_kernel<<<1, 32, 0, stream>>>(meta);
  // 3) LN + gating + routing (consumes xc)
  ln_route_kernel<<<T_TOK, 128, 0, stream>>>(xc, ln_g, ln_b, gate_w, xn, counts, lists, gates);
  // 4) weight conversion (independent; serialized on stream is fine)
  {
    size_t tot = (size_t)NE * DIM * HID;
    convert_w_kernel<<<(unsigned)((tot + 255) / 256), 256, 0, stream>>>(w1, w2, w1t, w2t);
  }
  // 5) zero accumulator (aliases xc — must follow ln_route)
  {
    size_t tot4 = (size_t)T_TOK * DIM / 4;
    zero_y_kernel<<<(unsigned)((tot4 + 255) / 256), 256, 0, stream>>>((float4*)y_acc);
  }
  // 6) build tile work list
  tile_build_kernel<<<1, 32, 0, stream>>>(counts, total, desc);
  // 7) fused MoE FFN (WMMA + async LDS gather)
  moe_kernel<<<MAX_TILES, 128, 0, stream>>>(xn, w1t, w2t, b1, b2, counts, total, desc,
                                            lists, gates, y_acc);
  // 8) residual + layer scale -> NCHW
  {
    size_t tot = (size_t)NB * DIM * HH * WW;
    combine_kernel<<<(unsigned)((tot + 255) / 256), 256, 0, stream>>>(input, lscale, y_acc, out);
  }
}
